// TCN_GCN_unit_25744033972349
// MI455X (gfx1250) — matrix operand order repaired and checked by reading, not compiled, not run on hardware
//
#include <hip/hip_runtime.h>
#include <hip/hip_bf16.h>

typedef __attribute__((ext_vector_type(16))) __bf16 v16bf;
typedef __attribute__((ext_vector_type(8)))  __bf16 v8bf;
typedef __attribute__((ext_vector_type(4)))  __bf16 v4bf;
typedef __attribute__((ext_vector_type(8)))  float  v8f;
typedef __attribute__((ext_vector_type(4)))  float  v4f;

#define NN 64
#define CC 64
#define TTOT 300
#define VV 25
#define SS 3
#define KW 9
#define TBLK 8                 // t-values per workgroup
#define COLS (TBLK*VV)         // 200 live columns (t,v)
#define CTILES 14              // padded so every wave owns exactly NCT tiles
#define COLP (CTILES*16)       // 224 padded columns
#define NCT 7                  // column tiles per wave (compile-time constant!)
#define CHW (TTOT*VV)          // 7500
#define ELEMS (NN*CC*CHW)      // 30,720,000
#define NSB 512                // stats partial blocks

#define AQ_N 3072              // 3*2*32*16
#define WD_N 12288             // 3*4*2*32*16
#define WT_N 36864             // 4*18*32*16

__device__ __forceinline__ float quant(float a) {
  float q = truncf(a * 1024.0f);
  q = fminf(fmaxf(q, -32768.0f), 32768.0f);
  return q * 0.0009765625f;
}
__device__ __forceinline__ __bf16 f2bf(float f) { return (__bf16)f; }

// A-operand (16x32 bf16) K index for slot e / lane
__device__ __forceinline__ int kmapA(int e, int lane) {
  return e + ((lane >= 16) ? 16 : 0);
}

// ---------------------------------------------------------------------------
// Prep 1: pack Aq (quantized), Wd, Wt into per-lane WMMA fragment order (bf16)
// ---------------------------------------------------------------------------
__global__ __launch_bounds__(256) void prep_frags(
    const float* __restrict__ A, const float* __restrict__ PA,
    const float* __restrict__ Wd, const float* __restrict__ Wt,
    __bf16* __restrict__ AqFrag, __bf16* __restrict__ WdFrag,
    __bf16* __restrict__ WtFrag)
{
  int i = blockIdx.x*256 + threadIdx.x;
  if (i < AQ_N) {
    // B-fragments of Aq: [s][nt][lane][e]
    int e = i & 15, lane = (i >> 4) & 31, nt = (i >> 9) & 1, s = i >> 10;
    int k = e + ((e >= 8) ? 8 : 0) + ((lane >= 16) ? 8 : 0);
    int coln = nt*16 + (lane & 15);
    float val = 0.0f;
    if (k < VV && coln < VV) {
      int idx = s*VV*VV + k*VV + coln;
      val = quant(A[idx] + PA[idx]);
    }
    AqFrag[i] = f2bf(val);
  } else if (i < AQ_N + WD_N) {
    // A-fragments of Wd: [s][mt][kstep][lane][e]
    int j = i - AQ_N;
    int e = j & 15, lane = (j >> 4) & 31, kstep = (j >> 9) & 1;
    int mt = (j >> 10) & 3, s = j >> 12;
    int o = mt*16 + (lane & 15);
    int cch = kstep*32 + kmapA(e, lane);
    WdFrag[j] = f2bf(Wd[(s*CC + o)*CC + cch]);
  } else if (i < AQ_N + WD_N + WT_N) {
    // A-fragments of Wt with kk-major K order (kappa = kk*64 + c): [mt][kstep][lane][e]
    int j = i - AQ_N - WD_N;
    int e = j & 15, lane = (j >> 4) & 31, r = j >> 9;
    int kstep = r % 18, mt = r / 18;
    int o = mt*16 + (lane & 15);
    int kk = kstep >> 1;
    int cch = ((kstep & 1) << 5) + kmapA(e, lane);
    WtFrag[j] = f2bf(Wt[(o*CC + cch)*KW + kk]);
  }
}

// ---------------------------------------------------------------------------
// Prep 2: x (NCTV) -> xcl (NTVC channel-last)
// ---------------------------------------------------------------------------
__global__ __launch_bounds__(256) void prep_xcl(
    const float* __restrict__ x, float* __restrict__ xcl)
{
  int i4 = blockIdx.x*256 + threadIdx.x;
  if (i4 >= ELEMS/4) return;
  int base = i4*4;
  int n = base / (CC*CHW);
  int rem = base - n*(CC*CHW);
  int c = rem / CHW;
  int tv = rem - c*CHW;
  v4f xv = *(const v4f*)&x[base];
  int ob = (n*CHW + tv)*CC + c;
  xcl[ob]      = xv[0];
  xcl[ob+CC]   = xv[1];
  xcl[ob+2*CC] = xv[2];
  xcl[ob+3*CC] = xv[3];
}

// ---------------------------------------------------------------------------
// Kernel 1: GCN  z = sum_s Wd[s] . (X . Aq[s]) + bias   -> z in NTVC layout
// ---------------------------------------------------------------------------
__global__ __launch_bounds__(256) void gcn_kernel(
    const float* __restrict__ x, const __bf16* __restrict__ AqFrag,
    const __bf16* __restrict__ WdFrag, const float* __restrict__ bd,
    float* __restrict__ z)
{
  __shared__ __bf16 sX[512][32];     // row = c*8+tl, K(v) padded 25->32 (zeros)
  __shared__ __bf16 sXAT[COLP][64];  // xa transposed: [col][channel]

  const int n    = blockIdx.y;
  const int t0   = blockIdx.x * TBLK;
  const int tid  = threadIdx.x;
  const int lane = tid & 31;
  const int wave = tid >> 5;

  for (int i = tid; i < 512*32; i += 256) {
    int row = i >> 5, k = i & 31;
    int c = row >> 3, tl = row & 7, t = t0 + tl;
    float val = (k < VV && t < TTOT) ? x[((n*CC + c)*TTOT + t)*VV + k] : 0.0f;
    sX[row][k] = f2bf(val);
  }
  for (int i = tid; i < (COLP - COLS)*64; i += 256)   // zero pad cols 200..223
    sXAT[COLS + (i >> 6)][i & 63] = f2bf(0.0f);
  __syncthreads();

  const int mt    = wave & 3;
  const int ct0   = wave >> 2;     // ct = ct0 + 2*i, i in [0, NCT)
  const int half8 = (lane >= 16) ? 8 : 0;
  const int kb    = (lane >= 16) ? 16 : 0;
  const v8f vzero = {0.f,0.f,0.f,0.f,0.f,0.f,0.f,0.f};
  v8f acc[NCT];
#pragma unroll
  for (int i = 0; i < NCT; ++i) acc[i] = vzero;

  for (int s = 0; s < SS; ++s) {
    // ---- xa = X * Aq[s] : rows (c*8+tl) x 32v, all-WMMA, no guards
    v8f cxa[4][2];
#pragma unroll
    for (int mi = 0; mi < 4; ++mi) {
      int row = (wave*4 + mi)*16 + (lane & 15);
      v8bf lo = *(const v8bf*)&sX[row][half8];
      v8bf hi = *(const v8bf*)&sX[row][half8 + 16];
      v16bf af = __builtin_shufflevector(lo, hi, 0,1,2,3,4,5,6,7,8,9,10,11,12,13,14,15);
#pragma unroll
      for (int nt = 0; nt < 2; ++nt) {
        v16bf bfr = *(const v16bf*)&AqFrag[((s*2 + nt)*32 + lane)*16];
        cxa[mi][nt] = __builtin_amdgcn_wmma_f32_16x16x32_bf16(
            false, af, false, bfr, (short)0, vzero, false, false);
      }
    }
    __syncthreads();   // previous subset's GEMM reads of sXAT complete
#pragma unroll
    for (int mi = 0; mi < 4; ++mi) {
#pragma unroll
      for (int nt = 0; nt < 2; ++nt) {
        int w = nt*16 + (lane & 15);
        if (w < VV) {
#pragma unroll
          for (int r = 0; r < 8; ++r) {
            int row = (wave*4 + mi)*16 + r + half8;
            sXAT[(row & 7)*VV + w][row >> 3] = f2bf(cxa[mi][nt][r]);
          }
        }
      }
    }
    __syncthreads();

    // ---- z += Wd[s] * xa : contiguous 32B fragment loads
#pragma unroll
    for (int kstep = 0; kstep < 2; ++kstep) {
      v16bf af = *(const v16bf*)&WdFrag[(((s*4 + mt)*2 + kstep)*32 + lane)*16];
#pragma unroll
      for (int i = 0; i < NCT; ++i) {
        int col = (ct0 + 2*i)*16 + (lane & 15);
        v16bf bfr = *(const v16bf*)&sXAT[col][kstep*32 + kb];
        acc[i] = __builtin_amdgcn_wmma_f32_16x16x32_bf16(
            false, af, false, bfr, (short)0, acc[i], false, false);
      }
    }
  }

  // store z (NTVC) + summed bias, two float4 stores per tile-column
  float bs[8];
#pragma unroll
  for (int r = 0; r < 8; ++r) {
    int o = mt*16 + half8 + r;
    bs[r] = bd[o] + bd[CC + o] + bd[2*CC + o];
  }
#pragma unroll
  for (int i = 0; i < NCT; ++i) {
    int col = (ct0 + 2*i)*16 + (lane & 15);
    if (col < COLS) {
      int tl = col / VV, v = col - tl*VV, t = t0 + tl;
      if (t < TTOT) {
        int base = ((n*TTOT + t)*VV + v)*CC + mt*16 + half8;
        v4f lo = {acc[i][0]+bs[0], acc[i][1]+bs[1], acc[i][2]+bs[2], acc[i][3]+bs[3]};
        v4f hi = {acc[i][4]+bs[4], acc[i][5]+bs[5], acc[i][6]+bs[6], acc[i][7]+bs[7]};
        *(v4f*)&z[base]     = lo;
        *(v4f*)&z[base + 4] = hi;
      }
    }
  }
}

// ---------------------------------------------------------------------------
// BN stats partial: NTVC input, per-lane fixed channels, float4 loads
// ---------------------------------------------------------------------------
__global__ __launch_bounds__(256) void stats_partial(
    const float* __restrict__ t, float* __restrict__ psum,
    float* __restrict__ psq)
{
  __shared__ float ls[256][4], lq[256][4];
  int tid = threadIdx.x;
  float s[4] = {0,0,0,0}, q[4] = {0,0,0,0};
  for (int i4 = blockIdx.x*256 + tid; i4 < ELEMS/4; i4 += NSB*256) {
    v4f v = *(const v4f*)&t[i4*4];
#pragma unroll
    for (int j = 0; j < 4; ++j) { s[j] += v[j]; q[j] += v[j]*v[j]; }
  }
#pragma unroll
  for (int j = 0; j < 4; ++j) { ls[tid][j] = s[j]; lq[tid][j] = q[j]; }
  __syncthreads();
  if (tid < 64) {   // channel c = tid ; held by threads with (t&15) == c/4, slot c&3
    int g = tid >> 2, j = tid & 3;
    float S = 0.f, Q = 0.f;
#pragma unroll
    for (int w = 0; w < 16; ++w) { S += ls[g + 16*w][j]; Q += lq[g + 16*w][j]; }
    psum[blockIdx.x*64 + tid] = S;
    psq [blockIdx.x*64 + tid] = Q;
  }
}

// finalize: fold BN into per-channel scale/shift
__global__ __launch_bounds__(64) void stats_finalize(
    const float* __restrict__ psum, const float* __restrict__ psq,
    const float* __restrict__ gamma, const float* __restrict__ beta,
    float* __restrict__ scale, float* __restrict__ shift)
{
  int c = threadIdx.x;
  float S = 0.f, Q = 0.f;
  for (int b = 0; b < NSB; ++b) { S += psum[b*64 + c]; Q += psq[b*64 + c]; }
  const float inv = 1.0f / (float)(NN*CHW);
  float m = S*inv;
  float var = Q*inv - m*m;
  float rs = rsqrtf(var + 1e-5f);
  float sc = rs * gamma[c];
  scale[c] = sc;
  shift[c] = beta[c] - m*sc;
}

// ---------------------------------------------------------------------------
// y = relu(BN1(z) + x), all NTVC, bf16 output
// ---------------------------------------------------------------------------
__global__ __launch_bounds__(256) void ybuild_kernel(
    const float* __restrict__ z, const float* __restrict__ xcl,
    const float* __restrict__ scale, const float* __restrict__ shift,
    __bf16* __restrict__ y)
{
  int i4 = blockIdx.x*256 + threadIdx.x;
  if (i4 >= ELEMS/4) return;
  int base = i4*4;
  int c0 = base & (CC - 1);
  v4f zv = *(const v4f*)&z[base];
  v4f xv = *(const v4f*)&xcl[base];
  v4bf o;
#pragma unroll
  for (int j = 0; j < 4; ++j) {
    float val = zv[j]*scale[c0+j] + shift[c0+j] + xv[j];
    o[j] = f2bf(fmaxf(val, 0.0f));
  }
  *(v4bf*)&y[base] = o;
}

// ---------------------------------------------------------------------------
// Kernel 4: 9x1 temporal conv, implicit im2col, kk-major K, NTVC in/out
// ---------------------------------------------------------------------------
__global__ __launch_bounds__(256) void tcn_kernel(
    const __bf16* __restrict__ y, const __bf16* __restrict__ WtFrag,
    const float* __restrict__ bt, float* __restrict__ conv)
{
  __shared__ __bf16 sYT[425][64];   // halo rows (t0-4..t0+11)*25+v, padded to 425

  const int n    = blockIdx.y;
  const int t0   = blockIdx.x * TBLK;
  const int tid  = threadIdx.x;
  const int lane = tid & 31;
  const int wave = tid >> 5;

  const v8bf zero8 = {f2bf(0.f),f2bf(0.f),f2bf(0.f),f2bf(0.f),
                      f2bf(0.f),f2bf(0.f),f2bf(0.f),f2bf(0.f)};
  for (int i = tid; i < 425*8; i += 256) {
    int row = i >> 3, c0 = (i & 7) * 8;
    v8bf val = zero8;
    if (row < 400) {
      int ts = row / VV, v = row - ts*VV, t = t0 + ts - 4;
      if (t >= 0 && t < TTOT)
        val = *(const v8bf*)&y[((n*TTOT + t)*VV + v)*CC + c0];
    }
    *(v8bf*)&sYT[row][c0] = val;
  }
  __syncthreads();

  const int mt    = wave & 3;
  const int ct0   = wave >> 2;
  const int half8 = (lane >= 16) ? 8 : 0;
  const int kb    = (lane >= 16) ? 16 : 0;
  const v8f vzero = {0.f,0.f,0.f,0.f,0.f,0.f,0.f,0.f};
  v8f acc[NCT];
#pragma unroll
  for (int i = 0; i < NCT; ++i) acc[i] = vzero;

  for (int kstep = 0; kstep < 18; ++kstep) {     // K = 9*64, kappa = kk*64 + c
    v16bf af = *(const v16bf*)&WtFrag[((mt*18 + kstep)*32 + lane)*16];
    int kk = kstep >> 1;
    int c0 = ((kstep & 1) << 5) + kb;
#pragma unroll
    for (int i = 0; i < NCT; ++i) {
      int col = (ct0 + 2*i)*16 + (lane & 15);
      int tl = col / VV, v = col - tl*VV;      // padded cols read valid rows, discarded at store
      v16bf bfr = *(const v16bf*)&sYT[(tl + kk)*VV + v][c0];
      acc[i] = __builtin_amdgcn_wmma_f32_16x16x32_bf16(
          false, af, false, bfr, (short)0, acc[i], false, false);
    }
  }

#pragma unroll
  for (int i = 0; i < NCT; ++i) {
    int col = (ct0 + 2*i)*16 + (lane & 15);
    if (col < COLS) {
      int tl = col / VV, v = col - tl*VV, t = t0 + tl;
      if (t < TTOT) {
        int base = ((n*TTOT + t)*VV + v)*CC + mt*16 + half8;
        int o0 = mt*16 + half8;
        v4f lo = {acc[i][0]+bt[o0],   acc[i][1]+bt[o0+1],
                  acc[i][2]+bt[o0+2], acc[i][3]+bt[o0+3]};
        v4f hi = {acc[i][4]+bt[o0+4], acc[i][5]+bt[o0+5],
                  acc[i][6]+bt[o0+6], acc[i][7]+bt[o0+7]};
        *(v4f*)&conv[base]     = lo;
        *(v4f*)&conv[base + 4] = hi;
      }
    }
  }
}

// ---------------------------------------------------------------------------
// out = relu(BN2(conv) + x)  (iterates NCTV: coalesced out/x, gathered conv)
// ---------------------------------------------------------------------------
__global__ __launch_bounds__(256) void final_kernel(
    const float* __restrict__ conv, const float* __restrict__ x,
    const float* __restrict__ scale, const float* __restrict__ shift,
    float* __restrict__ out)
{
  int i4 = blockIdx.x*256 + threadIdx.x;
  if (i4 >= ELEMS/4) return;
  int base = i4*4;
  int n = base / (CC*CHW);
  int rem = base - n*(CC*CHW);
  int c = rem / CHW;
  int tv = rem - c*CHW;
  int cb = (n*CHW + tv)*CC + c;
  v4f xv = *(const v4f*)&x[base];
  float sc = scale[c], sh = shift[c];
  v4f o;
#pragma unroll
  for (int j = 0; j < 4; ++j) {
    float val = conv[cb + j*CC]*sc + sh + xv[j];
    o[j] = fmaxf(val, 0.0f);
  }
  *(v4f*)&out[base] = o;
}

extern "C" void kernel_launch(void* const* d_in, const int* in_sizes, int n_in,
                              void* d_out, int out_size, void* d_ws, size_t ws_size,
                              hipStream_t stream) {
  const float* x      = (const float*)d_in[0];
  const float* A      = (const float*)d_in[1];
  const float* PA     = (const float*)d_in[2];
  const float* Wd     = (const float*)d_in[3];
  const float* bd     = (const float*)d_in[4];
  const float* gamma1 = (const float*)d_in[5];
  const float* beta1  = (const float*)d_in[6];
  const float* Wt     = (const float*)d_in[7];
  const float* bt     = (const float*)d_in[8];
  const float* gamma2 = (const float*)d_in[9];
  const float* beta2  = (const float*)d_in[10];
  float* out = (float*)d_out;

  // workspace layout
  float*  zbuf   = (float*)d_ws;                    // ELEMS f32 (z, then conv)
  float*  xcl    = zbuf + (size_t)ELEMS;            // ELEMS f32 channel-last x
  __bf16* ybuf   = (__bf16*)(xcl + (size_t)ELEMS);  // ELEMS bf16
  __bf16* AqFrag = ybuf + (size_t)ELEMS;
  __bf16* WdFrag = AqFrag + AQ_N;
  __bf16* WtFrag = WdFrag + WD_N;
  float*  psum   = (float*)(WtFrag + WT_N);
  float*  psq    = psum + NSB*64;
  float*  scale1 = psq + NSB*64;
  float*  shift1 = scale1 + 64;
  float*  scale2 = shift1 + 64;
  float*  shift2 = scale2 + 64;

  dim3 gemgrid((TTOT + TBLK - 1)/TBLK, NN);   // (38, 64)
  int vblocks = (ELEMS/4 + 255)/256;          // 30000
  int fblocks = (AQ_N + WD_N + WT_N + 255)/256;

  prep_frags   <<<fblocks, 256, 0, stream>>>(A, PA, Wd, Wt, AqFrag, WdFrag, WtFrag);
  prep_xcl     <<<vblocks, 256, 0, stream>>>(x, xcl);
  gcn_kernel   <<<gemgrid, 256, 0, stream>>>(x, AqFrag, WdFrag, bd, zbuf);
  stats_partial<<<NSB,     256, 0, stream>>>(zbuf, psum, psq);
  stats_finalize<<<1,       64, 0, stream>>>(psum, psq, gamma1, beta1, scale1, shift1);
  ybuild_kernel<<<vblocks, 256, 0, stream>>>(zbuf, xcl, scale1, shift1, ybuf);
  tcn_kernel   <<<gemgrid, 256, 0, stream>>>(ybuf, WtFrag, bt, zbuf);
  stats_partial<<<NSB,     256, 0, stream>>>(zbuf, psum, psq);
  stats_finalize<<<1,       64, 0, stream>>>(psum, psq, gamma2, beta2, scale2, shift2);
  final_kernel <<<vblocks, 256, 0, stream>>>(zbuf, x, scale2, shift2, out);
}
